// RefinerNet_54752243089489
// MI455X (gfx1250) — matrix operand order repaired and checked by reading, not compiled, not run on hardware
//
#include <hip/hip_runtime.h>
#include <math.h>

// ---------------- problem constants ----------------
#define BB    32
#define WW    96
#define NNODE 33
#define MM    100
#define FF    16
#define SDIM  66
#define HH    256
#define GHD   128
#define TIDD  182
#define TIDP  192          // TIDD padded to multiple of 32
#define NEDGE 64
#define RTOT (BB*WW*NNODE)   // 101376 = 64*1584
#define BWR  (BB*WW)         // 3072  = 64*48

typedef unsigned short u16;
typedef __attribute__((ext_vector_type(16))) __bf16 v16bf;
typedef __attribute__((ext_vector_type(8)))  float  v8f;

// float -> bf16 bits, round-to-nearest-even
static __device__ __forceinline__ u16 bfbits(float f) {
  union { float f; unsigned u; } x; x.f = f;
  unsigned r = x.u + 0x7fffu + ((x.u >> 16) & 1u);
  return (u16)(r >> 16);
}
static __device__ __forceinline__ unsigned bfpack(float a, float b) {
  return (unsigned)bfbits(a) | ((unsigned)bfbits(b) << 16);
}
static __device__ __forceinline__ float b2f(u16 h) {
  union { unsigned u; float f; } x; x.u = ((unsigned)h) << 16; return x.f;
}

static __device__ __forceinline__ v8f wmma_bf16(v16bf a, v16bf b, v8f c) {
  return __builtin_amdgcn_wmma_f32_16x16x32_bf16(false, a, false, b, (short)0, c,
                                                 false, false);
}
static __device__ __forceinline__ float sigm(float x) {
  return 1.0f / (1.0f + expf(-x));
}

// ---- packed fragment loads (contiguous-K layouts, k = 16h+s, 16B-aligned b128 pairs) ----
// A: lane holds row m, K = kb..kb+15 -> one 32B run
static __device__ __forceinline__ v16bf fragA(const u16* p /*&A[row][kb]*/) {
  union { v16bf v; uint4 q[2]; } u;
  const uint4* q = (const uint4*)p;
  u.q[0] = q[0];           // K kb..kb+7
  u.q[1] = q[1];           // K kb+8..kb+15 (+16 bytes)
  return u.v;
}
// B (BT layout, N x Kp): lane holds col n, K = kb..kb+15 -> one 32B run
static __device__ __forceinline__ v16bf fragB(const u16* p /*&BT[n][kb]*/) {
  union { v16bf v; uint4 q[2]; } u;
  const uint4* q = (const uint4*)p;
  u.q[0] = q[0]; u.q[1] = q[1];
  return u.v;
}
static __device__ __forceinline__ v16bf fragZ() {
  union { v16bf v; uint4 q[2]; } u;
  u.q[0] = uint4{0u,0u,0u,0u}; u.q[1] = uint4{0u,0u,0u,0u};
  return u.v;
}

// ---------------- weight conversion: (K x N) f32 -> (Npad x Kp) bf16 ----------
__global__ void k_convW(const float* __restrict__ W, u16* __restrict__ BT,
                        int K, int N, int Kp, int Npad) {
  int idx = blockIdx.x * 256 + threadIdx.x;
  if (idx >= Npad * Kp) return;
  int n = idx / Kp, k = idx - n * Kp;
  float v = (n < N && k < K) ? W[(size_t)k * N + n] : 0.f;
  BT[idx] = bfbits(v);
}
// weights already stored (N x K) f32 (Wih/Whh) -> (Npad x Kp) bf16
__global__ void k_convWT(const float* __restrict__ W, u16* __restrict__ BT,
                         int K, int N, int Kp, int Npad) {
  int idx = blockIdx.x * 256 + threadIdx.x;
  if (idx >= Npad * Kp) return;
  int n = idx / Kp, k = idx - n * Kp;
  float v = (n < N && k < K) ? W[(size_t)n * K + k] : 0.f;
  BT[idx] = bfbits(v);
}

// ---------------- graph setup: normalized padded neighbor table ----------------
// per node: up to 4 incoming edges (src index + normalized weight), zero-padded.
// Built serially by one thread -> deterministic fp accumulation order.
__global__ void k_graph(const int* __restrict__ esrc, const int* __restrict__ edst,
                        const float* __restrict__ ew,
                        int* __restrict__ nsrc, float* __restrict__ nwt) {
  __shared__ float deg[NNODE];
  __shared__ float dinv[NNODE];
  int t = threadIdx.x;
  if (t < NNODE) deg[t] = 0.f;
  __syncthreads();
  if (t < NEDGE) atomicAdd(&deg[edst[t]], ew[t]);
  __syncthreads();
  if (t < NNODE) dinv[t] = (deg[t] > 0.f) ? rsqrtf(deg[t]) : 0.f;
  __syncthreads();
  if (t == 0) {
    int cnt[NNODE];
    for (int n = 0; n < NNODE; ++n) {
      cnt[n] = 0;
      for (int j = 0; j < 4; ++j) { nsrc[n * 4 + j] = 0; nwt[n * 4 + j] = 0.f; }
    }
    for (int e = 0; e < NEDGE; ++e) {
      int d = edst[e], s = esrc[e];
      int j = cnt[d]++;
      if (j < 4) { nsrc[d * 4 + j] = s; nwt[d * 4 + j] = ew[e] * dinv[s] * dinv[d]; }
    }
  }
}

// ---------------- GNN pre-linear (K=18, VALU) ----------------
__global__ __launch_bounds__(128)
void k_pre(const float* __restrict__ feat, const float* __restrict__ state,
           const float* __restrict__ W, const float* __restrict__ bias,
           float* __restrict__ g) {
  int row = blockIdx.x;
  int c   = threadIdx.x;
  int bw  = row / NNODE;
  int n   = row - bw * NNODE;
  const float* f = feat + (size_t)bw * FF;
  float vm = state[(size_t)bw * SDIM + n];
  float va = state[(size_t)bw * SDIM + NNODE + n];
  float acc = bias[c];
#pragma unroll
  for (int i = 0; i < FF; ++i) acc += f[i] * W[i * GHD + c];
  acc += vm * W[FF * GHD + c] + va * W[(FF + 1) * GHD + c];
  g[(size_t)row * GHD + c] = acc;
}

// ---------------- fused TAG layer: prop chain in LDS + K=512 WMMA GEMM --------
// one block per (b,w); 8 waves; M-tiles 3 (48 rows, 33 valid), N-tiles 8 (=wave)
// static __shared__ (keeps ds_* addressing), branchless 4-way neighbor prop,
// bf16 LDS mirror feeding ds_load_b128 A-fragments.
__global__ __launch_bounds__(256)
void k_tag(const float* __restrict__ gin, float* __restrict__ gout,
           u16* __restrict__ goutb,                       // optional bf16 copy
           const u16* __restrict__ tagBT,                 // 4 chunks of (128x128) BT
           const float* __restrict__ tagb,
           const int* __restrict__ nsrc, const float* __restrict__ nwt) {
  __shared__ __align__(16) float sA[NNODE * GHD];
  __shared__ __align__(16) float sB[NNODE * GHD];
  __shared__ __align__(16) u16   sBF[NNODE * GHD];
  __shared__ int   ssrc[NNODE * 4];
  __shared__ float swt[NNODE * 4];

  int tid = threadIdx.x;
  int bw  = blockIdx.x;
  if (tid < NNODE * 4) { ssrc[tid] = nsrc[tid]; swt[tid] = nwt[tid]; }
  {
    const float4* g4 = (const float4*)(gin + (size_t)bw * NNODE * GHD);
    for (int i = tid; i < NNODE * GHD / 4; i += 256) {
      float4 v = g4[i];
      ((float4*)sA)[i] = v;
      ((uint2*)sBF)[i] = uint2{ bfpack(v.x, v.y), bfpack(v.z, v.w) };
    }
  }
  __syncthreads();

  int lane = tid & 31;
  int wave = tid >> 5;
  int mrow = lane & 15;
  int half = lane >> 4;
  int ncol = wave * 16 + (lane & 15);

  v8f acc[3];
#pragma unroll
  for (int mt = 0; mt < 3; ++mt)
#pragma unroll
    for (int i = 0; i < 8; ++i) acc[mt][i] = 0.f;

  for (int kk = 0; kk < 4; ++kk) {            // h0=g, h1, h2, h3
    if (kk > 0) {
      float* cur        = (kk & 1) ? sB : sA;
      const float* prev = (kk & 1) ? sA : sB;
      for (int i = tid; i < NNODE * (GHD / 2); i += 256) {   // column pairs
        int n = i >> 6, c0 = (i & 63) << 1;
        float s0 = 0.f, s1 = 0.f;
#pragma unroll
        for (int j = 0; j < 4; ++j) {
          float wv = swt[n * 4 + j];
          const float* pp = prev + ssrc[n * 4 + j] * GHD + c0;
          s0 += wv * pp[0];
          s1 += wv * pp[1];
        }
        cur[n * GHD + c0]     = s0;
        cur[n * GHD + c0 + 1] = s1;
        *(unsigned*)(sBF + n * GHD + c0) = bfpack(s0, s1);
      }
      __syncthreads();
    }
    const u16* Wc = tagBT + (size_t)kk * GHD * GHD;   // BT chunk (128 x 128)
    for (int ks = 0; ks < 4; ++ks) {
      int kbB = ks * 32 + half * 16;
      v16bf bf = fragB(Wc + (size_t)ncol * GHD + kbB);
      int kbA = ks * 32 + half * 16;
#pragma unroll
      for (int mt = 0; mt < 3; ++mt) {
        int row = mt * 16 + mrow;
        v16bf af = (row < NNODE) ? fragA(sBF + row * GHD + kbA) : fragZ();
        acc[mt] = wmma_bf16(af, bf, acc[mt]);
      }
    }
    __syncthreads();
  }

#pragma unroll
  for (int mt = 0; mt < 3; ++mt)
#pragma unroll
    for (int r = 0; r < 8; ++r) {
      int row = mt * 16 + r + 8 * half;
      if (row < NNODE) {
        float v = acc[mt][r] + tagb[ncol];
        v = v > 0.f ? v : 0.f;
        size_t o = ((size_t)bw * NNODE + row) * GHD + ncol;
        gout[o] = v;
        if (goutb) goutb[o] = bfbits(v);
      }
    }
}

// ---------------- generic bf16 WMMA GEMM, 64x64 block tile, 4 waves ----------
// A: bf16 (M x Kp) row-major, Kp % 32 == 0, M % 64 == 0 (always true here)
// BT: bf16 (Npad x Kp), pre-transposed + padded
// aMode 1: concat, k<256 from A, k>=256 from A2[bwmap(row/33)]
// epi: 0 bf16 store C; 1 fusion gate (C=u bf16); 2 dec2 split f32->eout; 3 f32->eout
static __device__ __forceinline__ int bwmap(int rbw) {
  return (rbw % WW) * BB + (rbw / WW);   // (b*96+w) -> (w*32+b)
}

__global__ __launch_bounds__(128)
void k_gemm(const u16* __restrict__ A, int lda,
            const u16* __restrict__ A2, int lda2, int aMode,
            int N, int Kp,
            const u16* __restrict__ BT, int ldk,
            const float* __restrict__ bias,
            u16* __restrict__ C, int ldc,
            int act, int epi,
            const u16* __restrict__ eg, const u16* __restrict__ es,
            float* __restrict__ eout) {
  int lane  = threadIdx.x & 31;
  int wave  = threadIdx.x >> 5;
  int mbase = blockIdx.x * 64;
  int nbase = blockIdx.y * 64 + wave * 16;
  int mrow  = lane & 15, half = lane >> 4;
  int ncol  = nbase + (lane & 15);

  v8f acc[4];
#pragma unroll
  for (int mt = 0; mt < 4; ++mt)
#pragma unroll
    for (int i = 0; i < 8; ++i) acc[mt][i] = 0.f;

  const u16* Brow = BT + (size_t)ncol * ldk;
  int ksteps = Kp >> 5;
  for (int ks = 0; ks < ksteps; ++ks) {
    int kbB = ks * 32 + half * 16;
    v16bf bf = fragB(Brow + kbB);
    if (ks + 1 < ksteps)
      __builtin_prefetch(Brow + kbB + 32, 0, 0);
    int kbA = ks * 32 + half * 16;
#pragma unroll
    for (int mt = 0; mt < 4; ++mt) {
      int row = mbase + mt * 16 + mrow;
      v16bf af;
      if (aMode == 1 && kbA >= 256)
        af = fragA(A2 + (size_t)bwmap(row / NNODE) * lda2 + (kbA - 256));
      else
        af = fragA(A + (size_t)row * lda + kbA);
      acc[mt] = wmma_bf16(af, bf, acc[mt]);
    }
  }

  if (ncol >= N) return;
  float bv = bias[ncol];
#pragma unroll
  for (int mt = 0; mt < 4; ++mt)
#pragma unroll
    for (int r = 0; r < 8; ++r) {
      int row = mbase + mt * 16 + r + 8 * half;
      float v = acc[mt][r] + bv;
      if (act == 1) v = v > 0.f ? v : 0.f;
      if (epi == 0) {
        C[(size_t)row * ldc + ncol] = bfbits(v);
      } else if (epi == 1) {                    // fusion: u = g*gn + (1-g)*s
        float gt = sigm(v);
        int a2r = bwmap(row / NNODE);
        float u = gt * b2f(eg[(size_t)row * HH + ncol]) +
                  (1.f - gt) * b2f(es[(size_t)a2r * HH + ncol]);
        C[(size_t)row * ldc + ncol] = bfbits(u);
      } else if (epi == 2) {                    // dec2 split write (f32 final out)
        int rbw = row / NNODE, nn = row - rbw * NNODE;
        if (ncol == 0)      eout[(size_t)rbw * (2 * NNODE) + nn] = v;
        else if (ncol == 1) eout[(size_t)rbw * (2 * NNODE) + NNODE + nn] = v;
      } else {                                   // epi 3: f32 store (gi preacts)
        eout[(size_t)row * ldc + ncol] = v;
      }
    }
}

// ---------------- gru_in builder (bf16, K padded 182->192) ----------------
__global__ void k_build_gruin(const float* __restrict__ z, const float* __restrict__ st,
                              const float* __restrict__ ft, u16* __restrict__ gin) {
  int idx = blockIdx.x * blockDim.x + threadIdx.x;
  if (idx >= BWR * TIDP) return;
  int r = idx / TIDP, i = idx - r * TIDP;
  int b = r & (BB - 1), w = r >> 5;            // row order: w*32+b (seq-major)
  int bw = b * WW + w;
  float v = 0.f;
  if (i < MM)                 v = z[(size_t)bw * MM + i];
  else if (i < MM + SDIM)     v = st[(size_t)bw * SDIM + (i - MM)];
  else if (i < TIDD)          v = ft[(size_t)bw * FF + (i - MM - SDIM)];
  gin[idx] = bfbits(v);
}

__global__ void k_zero32(float* p, int n) {
  int i = blockIdx.x * blockDim.x + threadIdx.x;
  if (i < n) p[i] = 0.f;
}
__global__ void k_zero16(u16* p, int n) {
  int i = blockIdx.x * blockDim.x + threadIdx.x;
  if (i < n) p[i] = 0;
}

// ---------------- GRU recurrence step ------------------------------------
// grid 16 (n-slices of 16), block 32 (1 wave). 6 accum tiles: 2 M x 3 gates.
__global__ __launch_bounds__(32)
void k_gru_step(const float* __restrict__ gi,     // 32x768 f32 (includes bih)
                const float* __restrict__ hinf,   // 32x256 f32
                const u16*   __restrict__ hinb,   // 32x256 bf16 mirror
                float* __restrict__ houtf,
                u16*   __restrict__ houtb,
                const u16* __restrict__ WhhBT,    // 768x256 bf16 (N x K)
                const float* __restrict__ bhh,
                u16* __restrict__ y) {            // 32x256 bf16 output slice
  int lane = threadIdx.x & 31;
  int mrow = lane & 15, half = lane >> 4;
  int j0 = blockIdx.x * 16;

  v8f acc[2][3];
#pragma unroll
  for (int mt = 0; mt < 2; ++mt)
#pragma unroll
    for (int g = 0; g < 3; ++g)
#pragma unroll
      for (int i = 0; i < 8; ++i) acc[mt][g][i] = 0.f;

#pragma unroll
  for (int ks = 0; ks < 8; ++ks) {
    int kbA = ks * 32 + half * 16;
    v16bf af0 = fragA(hinb + (0 * 16 + mrow) * HH + kbA);
    v16bf af1 = fragA(hinb + (1 * 16 + mrow) * HH + kbA);
    int kbB = ks * 32 + half * 16;
#pragma unroll
    for (int g = 0; g < 3; ++g) {
      int wr = g * HH + j0 + (lane & 15);
      v16bf bf = fragB(WhhBT + (size_t)wr * HH + kbB);
      acc[0][g] = wmma_bf16(af0, bf, acc[0][g]);
      acc[1][g] = wmma_bf16(af1, bf, acc[1][g]);
    }
  }

  int j = j0 + (lane & 15);
#pragma unroll
  for (int mt = 0; mt < 2; ++mt)
#pragma unroll
    for (int r = 0; r < 8; ++r) {
      int b = mt * 16 + r + 8 * half;
      float ghr = acc[mt][0][r] + bhh[j];
      float ghz = acc[mt][1][r] + bhh[HH + j];
      float ghn = acc[mt][2][r] + bhh[2 * HH + j];
      const float* gib = gi + (size_t)b * 768;
      float rr = sigm(gib[j] + ghr);
      float zz = sigm(gib[HH + j] + ghz);
      float nn = tanhf(gib[2 * HH + j] + rr * ghn);
      float hv = (1.f - zz) * nn + zz * hinf[b * HH + j];
      houtf[b * HH + j] = hv;
      u16 hb = bfbits(hv);
      houtb[b * HH + j] = hb;
      y[b * HH + j]     = hb;
    }
}

// ---------------- host orchestration ----------------
extern "C" void kernel_launch(void* const* d_in, const int* in_sizes, int n_in,
                              void* d_out, int out_size, void* d_ws, size_t ws_size,
                              hipStream_t stream) {
  const float* state   = (const float*)d_in[0];
  const float* z_seq   = (const float*)d_in[1];
  const float* feat    = (const float*)d_in[2];
  const int*   esrc    = (const int*)d_in[3];
  const int*   edst    = (const int*)d_in[4];
  const float* eweight = (const float*)d_in[5];
  const float* pre_W   = (const float*)d_in[6];
  const float* pre_b   = (const float*)d_in[7];
  const float* tag0W   = (const float*)d_in[8];
  const float* tag0b   = (const float*)d_in[9];
  const float* tag1W   = (const float*)d_in[10];
  const float* tag1b   = (const float*)d_in[11];
  const float* post_W  = (const float*)d_in[12];
  const float* post_b  = (const float*)d_in[13];
  const float* tpre_W1 = (const float*)d_in[14];
  const float* tpre_b1 = (const float*)d_in[15];
  const float* tpre_W2 = (const float*)d_in[16];
  const float* tpre_b2 = (const float*)d_in[17];
  const float* gru_Wih = (const float*)d_in[18];
  const float* gru_Whh = (const float*)d_in[19];
  const float* gru_bih = (const float*)d_in[20];
  const float* gru_bhh = (const float*)d_in[21];
  const float* fus_W   = (const float*)d_in[22];
  const float* fus_b   = (const float*)d_in[23];
  const float* dec_W1  = (const float*)d_in[24];
  const float* dec_b1  = (const float*)d_in[25];
  const float* dec_W2  = (const float*)d_in[26];
  const float* dec_b2  = (const float*)d_in[27];

  char* wp = (char*)d_ws;
  auto alloc = [&](size_t bytes) -> char* {
    char* p = wp;
    wp += (bytes + 255) & ~(size_t)255;
    return p;
  };
  // graph tables
  int*   NSRC = (int*)alloc(NNODE * 4 * 4);
  float* NWT  = (float*)alloc(NNODE * 4 * 4);
  // f32 buffers
  float* gA  = (float*)alloc((size_t)RTOT * GHD * 4);
  float* gB  = (float*)alloc((size_t)RTOT * GHD * 4);
  float* GI  = (float*)alloc((size_t)BWR * 768 * 4);
  float* H0f = (float*)alloc((size_t)BB * HH * 4);
  float* H1f = (float*)alloc((size_t)BB * HH * 4);
  // bf16 activation buffers
  u16* Gb   = (u16*)alloc((size_t)RTOT * GHD * 2);   // TAG output for post GEMM
  u16* GNb  = (u16*)alloc((size_t)RTOT * HH * 2);    // g_nodes; later dec1 out
  u16* R1b  = (u16*)alloc((size_t)RTOT * HH * 2);    // fusion u
  u16* GINb = (u16*)alloc((size_t)BWR * TIDP * 2);
  u16* S1b  = (u16*)alloc((size_t)BWR * HH * 2);
  u16* S2b  = (u16*)alloc((size_t)BWR * HH * 2);
  u16* Y0b  = (u16*)alloc((size_t)BWR * HH * 2);
  u16* Y1b  = (u16*)alloc((size_t)BWR * HH * 2);
  u16* H0b  = (u16*)alloc((size_t)BB * HH * 2);
  u16* H1b  = (u16*)alloc((size_t)BB * HH * 2);
  // bf16 transposed+padded weights (Npad x Kp)
  u16* postBT  = (u16*)alloc((size_t)256 * 128 * 2);
  u16* tp1BT   = (u16*)alloc((size_t)256 * TIDP * 2);
  u16* tp2BT   = (u16*)alloc((size_t)256 * 256 * 2);
  u16* wihBT0  = (u16*)alloc((size_t)768 * 256 * 2);
  u16* wihBT1  = (u16*)alloc((size_t)768 * 256 * 2);
  u16* whhBT0  = (u16*)alloc((size_t)768 * 256 * 2);
  u16* whhBT1  = (u16*)alloc((size_t)768 * 256 * 2);
  u16* fusBT   = (u16*)alloc((size_t)256 * 512 * 2);
  u16* dc1BT   = (u16*)alloc((size_t)256 * 256 * 2);
  u16* dc2BT   = (u16*)alloc((size_t)64 * 256 * 2);
  u16* tagBT0  = (u16*)alloc((size_t)4 * 128 * 128 * 2);
  u16* tagBT1  = (u16*)alloc((size_t)4 * 128 * 128 * 2);

  auto convW = [&](const float* W, u16* BT, int K, int N, int Kp, int Npad) {
    int tot = Npad * Kp;
    k_convW<<<(tot + 255) / 256, 256, 0, stream>>>(W, BT, K, N, Kp, Npad);
  };
  auto convWT = [&](const float* W, u16* BT, int K, int N, int Kp, int Npad) {
    int tot = Npad * Kp;
    k_convWT<<<(tot + 255) / 256, 256, 0, stream>>>(W, BT, K, N, Kp, Npad);
  };

  // 0) one-time weight conversion (bf16, transposed, padded)
  convW(post_W,  postBT, 128, 256, 128, 256);
  convW(tpre_W1, tp1BT,  TIDD, 256, TIDP, 256);
  convW(tpre_W2, tp2BT,  256, 256, 256, 256);
  convWT(gru_Wih,                   wihBT0, 256, 768, 256, 768);
  convWT(gru_Wih + (size_t)768*256, wihBT1, 256, 768, 256, 768);
  convWT(gru_Whh,                   whhBT0, 256, 768, 256, 768);
  convWT(gru_Whh + (size_t)768*256, whhBT1, 256, 768, 256, 768);
  convW(fus_W,  fusBT, 512, 256, 512, 256);
  convW(dec_W1, dc1BT, 256, 256, 256, 256);
  convW(dec_W2, dc2BT, 256, 2, 256, 64);
  for (int kk = 0; kk < 4; ++kk) {
    convW(tag0W + (size_t)kk * 128 * 128, tagBT0 + (size_t)kk * 128 * 128, 128, 128, 128, 128);
    convW(tag1W + (size_t)kk * 128 * 128, tagBT1 + (size_t)kk * 128 * 128, 128, 128, 128, 128);
  }

  // 1) graph normalization + padded neighbor table
  k_graph<<<1, 64, 0, stream>>>(esrc, edst, eweight, NSRC, NWT);

  // 2) GNN pre-linear
  k_pre<<<RTOT, 128, 0, stream>>>(feat, state, pre_W, pre_b, gA);

  // 3) TAG layers (fused prop + K=512 WMMA GEMM); layer 1 emits bf16 copy
  k_tag<<<BWR, 256, 0, stream>>>(gA, gB, nullptr, tagBT0, tag0b, NSRC, NWT);
  k_tag<<<BWR, 256, 0, stream>>>(gB, gA, Gb,      tagBT1, tag1b, NSRC, NWT);

  // 4) post: g_nodes = g @ post_W + post_b  (bf16 out)
  {
    dim3 g(RTOT / 64, 4);
    k_gemm<<<g, 128, 0, stream>>>(Gb, GHD, nullptr, 0, 0,
                                  HH, GHD, postBT, GHD, post_b,
                                  GNb, HH, 0, 0, nullptr, nullptr, nullptr);
  }

  // 5) temporal pre-net (rows in (w*32+b) seq-major order)
  {
    int tot = BWR * TIDP;
    k_build_gruin<<<(tot + 255) / 256, 256, 0, stream>>>(z_seq, state, feat, GINb);
    dim3 g(BWR / 64, 4);
    k_gemm<<<g, 128, 0, stream>>>(GINb, TIDP, nullptr, 0, 0,
                                  HH, TIDP, tp1BT, TIDP, tpre_b1,
                                  S1b, HH, 1, 0, nullptr, nullptr, nullptr);
    k_gemm<<<g, 128, 0, stream>>>(S1b, HH, nullptr, 0, 0,
                                  HH, HH, tp2BT, HH, tpre_b2,
                                  S2b, HH, 1, 0, nullptr, nullptr, nullptr);
  }

  // 6) GRU: per layer, one time-parallel input GEMM + 96 recurrence steps
  const u16* xin[2]   = { S2b, Y0b };
  u16* yout[2]        = { Y0b, Y1b };
  const u16* wihBT[2] = { wihBT0, wihBT1 };
  const u16* whhBT[2] = { whhBT0, whhBT1 };
  for (int l = 0; l < 2; ++l) {
    dim3 g(BWR / 64, 12);
    k_gemm<<<g, 128, 0, stream>>>(xin[l], HH, nullptr, 0, 0,
                                  3 * HH, HH, wihBT[l], HH,
                                  gru_bih + (size_t)l * 768,
                                  nullptr, 3 * HH, 0, 3, nullptr, nullptr, GI);
    k_zero32<<<(BB * HH + 255) / 256, 256, 0, stream>>>(H0f, BB * HH);
    k_zero16<<<(BB * HH + 255) / 256, 256, 0, stream>>>(H0b, BB * HH);
    float* hinf = H0f; float* houtf = H1f;
    u16*   hinb = H0b; u16*   houtb = H1b;
    for (int t = 0; t < WW; ++t) {
      k_gru_step<<<16, 32, 0, stream>>>(GI + (size_t)t * BB * 768,
                                        hinf, hinb, houtf, houtb,
                                        whhBT[l], gru_bhh + (size_t)l * 768,
                                        yout[l] + (size_t)t * BB * HH);
      float* tf = hinf; hinf = houtf; houtf = tf;
      u16*   tb = hinb; hinb = houtb; houtb = tb;
    }
  }

  // 7) fusion gate: u = sig([g_nodes|s_t]@fus_W+b)*gn + (1-sig)*s_t -> R1b
  {
    dim3 g(RTOT / 64, 4);
    k_gemm<<<g, 128, 0, stream>>>(GNb, HH, Y1b, HH, 1,
                                  HH, 2 * HH, fusBT, 2 * HH, fus_b,
                                  R1b, HH, 0, 1, GNb, Y1b, nullptr);
  }

  // 8) decoder
  {
    dim3 g(RTOT / 64, 4);
    k_gemm<<<g, 128, 0, stream>>>(R1b, HH, nullptr, 0, 0,
                                  HH, HH, dc1BT, HH, dec_b1,
                                  GNb, HH, 1, 0, nullptr, nullptr, nullptr);
    dim3 g2(RTOT / 64, 1);
    k_gemm<<<g2, 128, 0, stream>>>(GNb, HH, nullptr, 0, 0,
                                   2, HH, dc2BT, HH, dec_b2,
                                   nullptr, 0, 0, 2, nullptr, nullptr,
                                   (float*)d_out);
  }

  (void)in_sizes; (void)n_in; (void)out_size; (void)ws_size;
}
